// DeepRTE_24592982737602
// MI455X (gfx1250) — matrix operand order repaired and checked by reading, not compiled, not run on hardware
//
#include <hip/hip_runtime.h>

// ---------------------------------------------------------------------------
// DeepRTE forward for MI455X (gfx1250): wave32, WMMA bf16 16x16x32 (f32 acc)
// for the three MLP stacks, WMMA f32 16x16x4 for the rw_vs contraction.
// ---------------------------------------------------------------------------

typedef __attribute__((ext_vector_type(16))) __bf16 v16bf;
typedef __attribute__((ext_vector_type(8)))  __bf16 v8bf;
typedef __attribute__((ext_vector_type(8)))  float  v8f;
typedef __attribute__((ext_vector_type(2)))  float  v2f;

#define DB 4
#define DN 32
#define DM 64
#define DK 16
#define DK1 17
#define DP 128
#define WW 128

// Branchless tanh via hardware exp: tanh(x) = (e^2x - 1)/(e^2x + 1).
// Clamp keeps e^2x finite (avoids inf/inf); |x|>=15 saturates in f32 anyway.
__device__ __forceinline__ float fast_tanh(float x) {
  x = fminf(fmaxf(x, -15.f), 15.f);
  const float e = __expf(2.f * x);
  return (e - 1.f) / (e + 1.f);
}

// ---- WMMA fragment loaders (layouts per CDNA5 ISA §7.12.2) ------
// A (16x32 bf16): lane l -> row (l&15); lanes<16 hold K 0..7 & 16..23,
// lanes>=16 hold K 8..15 & 24..31.  Two 16B loads per lane.
__device__ __forceinline__ v16bf ld_afrag(const __bf16* H, int ld, int r0,
                                          int kc, int lane) {
  const int r    = r0 + (lane & 15);
  const int base = (lane < 16) ? 0 : 8;
  const __bf16* p = H + r * ld + kc + base;
  v8bf lo = *(const v8bf*)(p);
  v8bf hi = *(const v8bf*)(p + 16);
  v16bf a;
#pragma unroll
  for (int e = 0; e < 8; ++e) { a[e] = lo[e]; a[e + 8] = hi[e]; }
  return a;
}

// B (32x16 bf16): lane l -> col (l&15); lanes<16 hold K 0..7 & 16..23,
// lanes>=16 hold K 8..15 & 24..31.  Wt stored [out][in] row-major.
__device__ __forceinline__ v16bf ld_bfrag(const __bf16* Wt, int ld, int nc,
                                          int kc, int lane) {
  const int n  = nc + (lane & 15);
  const int kb = (lane < 16) ? 0 : 8;
  const __bf16* p = Wt + n * ld + kc + kb;
  v8bf lo = *(const v8bf*)(p);
  v8bf hi = *(const v8bf*)(p + 16);
  v16bf b;
#pragma unroll
  for (int e = 0; e < 8; ++e) { b[e] = lo[e]; b[e + 8] = hi[e]; }
  return b;
}

__device__ __forceinline__ v8f wmma_bf16(v16bf a, v16bf b, v8f c) {
  return __builtin_amdgcn_wmma_f32_16x16x32_bf16(false, a, false, b,
                                                 (short)0, c, false, false);
}

// ---------------------------------------------------------------------------
// Kernel 0: bf16 copy of sw0 (WMMA B operand, already [out][in] layout).
// ---------------------------------------------------------------------------
__global__ void k_prep(const float* __restrict__ sw0, __bf16* __restrict__ sw0b) {
  int i = blockIdx.x * blockDim.x + threadIdx.x;
  if (i < 128 * 128) sw0b[i] = (__bf16)sw0[i];
}

// ---------------------------------------------------------------------------
// Kernel 1: attention MLP (6->64->64->1), softmax over P, coeff.
// One block per (b,n,k1); 8 waves, each owns a 16-row (p) tile.
// ---------------------------------------------------------------------------
__global__ __launch_bounds__(256) void k_attn(
    const float* __restrict__ phase, const float* __restrict__ posc,
    const float* __restrict__ sigma, const float* __restrict__ velc,
    const float* __restrict__ aw0, const float* __restrict__ ab0,
    const float* __restrict__ aw1, const float* __restrict__ ab1,
    const float* __restrict__ aw2, const float* __restrict__ ab2,
    float* __restrict__ coeff) {
  __shared__ __align__(16) __bf16 aw0t[64 * 40];   // [out64][in pad 40]
  __shared__ __align__(16) __bf16 aw1t[64 * 72];   // [out64][in pad 72]
  __shared__ float aw2s[64], ab0s[64], ab1s[64];
  __shared__ __align__(16) __bf16 Xf[128 * 40];    // features, pad ld 40
  __shared__ __align__(16) __bf16 H1[8][16 * 72];
  __shared__ __align__(16) __bf16 H2[8][16 * 72];
  __shared__ float ploc[128], lgs[128], red[128];

  const int tid = threadIdx.x;
  const int idx = blockIdx.x;
  const int kk  = idx % DK1;
  const int n   = (idx / DK1) % DN;
  const int b   = idx / (DK1 * DN);

  for (int i = tid; i < 64 * 40; i += 256) {
    int o = i / 40, c = i % 40;
    aw0t[i] = (c < 6) ? (__bf16)aw0[c * 64 + o] : (__bf16)0.f;
  }
  for (int i = tid; i < 64 * 72; i += 256) {
    int o = i / 72, c = i % 72;
    aw1t[i] = (c < 64) ? (__bf16)aw1[c * 64 + o] : (__bf16)0.f;
  }
  if (tid < 64) { aw2s[tid] = aw2[tid]; ab0s[tid] = ab0[tid]; ab1s[tid] = ab1[tid]; }

  const float x0 = phase[(b * DN + n) * 4 + 0];
  const float x1 = phase[(b * DN + n) * 4 + 1];
  float va0, va1;
  if (kk == 0) { va0 = phase[(b*DN+n)*4+2]; va1 = phase[(b*DN+n)*4+3]; }
  else         { va0 = velc[(b*DK+kk-1)*2+0]; va1 = velc[(b*DK+kk-1)*2+1]; }
  const float vs  = sqrtf(va0*va0 + va1*va1 + 1e-16f);
  const float ag0 = va0 / vs, ag1 = va1 / vs;

  if (tid < 128) {
    const int p = tid;
    const float r0 = x0 - posc[(b*DP+p)*2+0];
    const float r1 = x1 - posc[(b*DP+p)*2+1];
    const float rd = sqrtf(r0*r0 + r1*r1 + 1e-16f);
    const float pl = r0*ag0 + r1*ag1;
    const float al = pl / (rd + 1e-8f);
    ploc[p] = pl;
    __bf16* row = &Xf[p * 40];
    row[0]=(__bf16)x0; row[1]=(__bf16)x1; row[2]=(__bf16)va0; row[3]=(__bf16)va1;
    row[4]=(__bf16)al; row[5]=(__bf16)pl;
#pragma unroll
    for (int c = 6; c < 40; ++c) row[c] = (__bf16)0.f;
  }
  __syncthreads();

  const int wv = tid >> 5, lane = tid & 31;
  const int r0row = wv * 16;
  __bf16* h1 = H1[wv];
  __bf16* h2 = H2[wv];

  // layer 0: 32(pad) -> 64
  {
    v16bf a0 = ld_afrag(Xf, 40, r0row, 0, lane);
    for (int nt = 0; nt < 4; ++nt) {
      v8f c = {0,0,0,0,0,0,0,0};
      c = wmma_bf16(a0, ld_bfrag(aw0t, 40, nt * 16, 0, lane), c);
      const int col = nt * 16 + (lane & 15);
      const int mb  = (lane < 16) ? 0 : 8;
#pragma unroll
      for (int j = 0; j < 8; ++j)
        h1[(j + mb) * 72 + col] = (__bf16)fast_tanh(c[j] + ab0s[col]);
    }
  }
  // layer 1: 64 -> 64
  for (int nt = 0; nt < 4; ++nt) {
    v8f c = {0,0,0,0,0,0,0,0};
#pragma unroll
    for (int ks = 0; ks < 2; ++ks)
      c = wmma_bf16(ld_afrag(h1, 72, 0, ks * 32, lane),
                    ld_bfrag(aw1t, 72, nt * 16, ks * 32, lane), c);
    const int col = nt * 16 + (lane & 15);
    const int mb  = (lane < 16) ? 0 : 8;
#pragma unroll
    for (int j = 0; j < 8; ++j)
      h2[(j + mb) * 72 + col] = (__bf16)fast_tanh(c[j] + ab1s[col]);
  }
  // layer 2: 64 -> 1 (VALU dot), mask
  if (lane < 16) {
    const int p = r0row + lane;
    float acc = ab2[0];
#pragma unroll
    for (int w = 0; w < 64; ++w) acc += (float)h2[lane * 72 + w] * aw2s[w];
    lgs[p] = (ploc[p] > 0.f) ? acc : -1e30f;
  }
  __syncthreads();

  // softmax over p=0..127
  if (tid < 128) red[tid] = lgs[tid];
  __syncthreads();
  for (int s = 64; s > 0; s >>= 1) {
    if (tid < s) red[tid] = fmaxf(red[tid], red[tid + s]);
    __syncthreads();
  }
  const float mx = red[0];
  __syncthreads();
  float ev = 0.f;
  if (tid < 128) { ev = __expf(lgs[tid] - mx); red[tid] = ev; }
  __syncthreads();
  for (int s = 64; s > 0; s >>= 1) {
    if (tid < s) red[tid] += red[tid + s];
    __syncthreads();
  }
  const float tot = red[0];
  __syncthreads();
  const float aw_ = (tid < 128) ? (ev / tot) : 0.f;

  float sc[2];
#pragma unroll
  for (int ch = 0; ch < 2; ++ch) {
    if (tid < 128) red[tid] = aw_ * sigma[(b * DP + tid) * 2 + ch];
    __syncthreads();
    for (int s = 64; s > 0; s >>= 1) {
      if (tid < s) red[tid] += red[tid + s];
      __syncthreads();
    }
    sc[ch] = red[0];
    __syncthreads();
  }
  if (tid == 0) {
    coeff[idx * 2 + 0] = __expf(-sc[0]);
    coeff[idx * 2 + 1] = __expf(-sc[1]);
  }
}

// ---------------------------------------------------------------------------
// Kernel 2: transport MLP (10->256->256->128, tanh all, exp).  Persistent
// blocks: all three weight matrices transposed+padded to bf16 in LDS
// (~293KB of the 320KB WGP LDS).  256 threads = 8 waves: wave PAIRS share a
// 16-row H tile and split the output n-tiles, giving 2 waves/SIMD32 for
// latency hiding (vs 1 with the 4-wave variant).
// ---------------------------------------------------------------------------
__global__ __launch_bounds__(256) void k_tmlp(
    const float* __restrict__ phase, const float* __restrict__ bco,
    const float* __restrict__ velc, const float* __restrict__ coeff,
    const float* __restrict__ tw0, const float* __restrict__ tb0,
    const float* __restrict__ tw1, const float* __restrict__ tb1,
    const float* __restrict__ tw2, const float* __restrict__ tb2,
    float* __restrict__ g) {
  __shared__ __align__(16) __bf16 tw0t[256 * 40];
  __shared__ __align__(16) __bf16 tw1t[256 * 264];
  __shared__ __align__(16) __bf16 tw2t[128 * 264];
  __shared__ float tb0s[256], tb1s[256], tb2s[128];
  __shared__ __align__(16) __bf16 H1[4][16 * 264];
  __shared__ __align__(16) __bf16 H2[4][16 * 264];

  const int tid = threadIdx.x;
  for (int i = tid; i < 256 * 40; i += 256) {
    int o = i / 40, c = i % 40;
    tw0t[i] = (c < 10) ? (__bf16)tw0[c * 256 + o] : (__bf16)0.f;
  }
  for (int i = tid; i < 256 * 264; i += 256) {
    int o = i / 264, c = i % 264;
    tw1t[i] = (c < 256) ? (__bf16)tw1[c * 256 + o] : (__bf16)0.f;
  }
  for (int i = tid; i < 128 * 264; i += 256) {
    int o = i / 264, c = i % 264;
    tw2t[i] = (c < 256) ? (__bf16)tw2[c * 128 + o] : (__bf16)0.f;
  }
  if (tid < 256) { tb0s[tid] = tb0[tid]; tb1s[tid] = tb1[tid]; }
  if (tid < 128) tb2s[tid] = tb2[tid];
  __syncthreads();

  const int wv   = tid >> 5, lane = tid & 31;
  const int pair = wv >> 1;   // 0..3 : which 16-row tile
  const int sub  = wv & 1;    // 0..1 : which half of the n-tiles
  __bf16* h1 = H1[pair];
  __bf16* h2 = H2[pair];
  const int nchunk = (DB * DN * DM * DK1) / 64;  // 2176

  for (int chunk = blockIdx.x; chunk < nchunk; chunk += gridDim.x) {
    const int rowbase = chunk * 64 + pair * 16;

    // Build the 16x32 A fragment for layer 0 straight from inputs.
    const int r   = lane & 15;
    const int row = rowbase + r;
    const int kk  = row % DK1;
    const int m   = (row / DK1) % DM;
    const int n   = (row / (DK1 * DM)) % DN;
    const int b   = row / (DK1 * DM * DN);
    const int base = (lane < 16) ? 0 : 8;
    v16bf a0;
#pragma unroll
    for (int e = 0; e < 16; ++e) {
      const int k = base + ((e < 8) ? e : e + 8);
      float v = 0.f;
      if (k < 2)       v = phase[(b * DN + n) * 4 + k];
      else if (k < 4)  v = (kk == 0) ? phase[(b * DN + n) * 4 + k]
                                     : velc[(b * DK + kk - 1) * 2 + (k - 2)];
      else if (k < 8)  v = bco[(b * DM + m) * 4 + (k - 4)];
      else if (k < 10) v = coeff[((b * DN + n) * DK1 + kk) * 2 + (k - 8)];
      a0[e] = (__bf16)v;
    }

    // layer 0: 32(pad) -> 256 ; this wave does 8 of the 16 n-tiles
    for (int t = 0; t < 8; ++t) {
      const int nt = sub * 8 + t;
      v8f c = {0,0,0,0,0,0,0,0};
      c = wmma_bf16(a0, ld_bfrag(tw0t, 40, nt * 16, 0, lane), c);
      const int col = nt * 16 + (lane & 15);
      const int mb  = (lane < 16) ? 0 : 8;
#pragma unroll
      for (int j = 0; j < 8; ++j)
        h1[(j + mb) * 264 + col] = (__bf16)fast_tanh(c[j] + tb0s[col]);
    }
    __syncthreads();   // partner wave's H1 columns are now visible

    // layer 1: 256 -> 256 ; 8 of 16 n-tiles
    for (int t = 0; t < 8; ++t) {
      const int nt = sub * 8 + t;
      v8f c = {0,0,0,0,0,0,0,0};
#pragma unroll
      for (int ks = 0; ks < 8; ++ks)
        c = wmma_bf16(ld_afrag(h1, 264, 0, ks * 32, lane),
                      ld_bfrag(tw1t, 264, nt * 16, ks * 32, lane), c);
      const int col = nt * 16 + (lane & 15);
      const int mb  = (lane < 16) ? 0 : 8;
#pragma unroll
      for (int j = 0; j < 8; ++j)
        h2[(j + mb) * 264 + col] = (__bf16)fast_tanh(c[j] + tb1s[col]);
    }
    __syncthreads();   // partner wave's H2 columns are now visible

    // layer 2: 256 -> 128, tanh, exp, store g ; 4 of 8 n-tiles
    for (int t = 0; t < 4; ++t) {
      const int nt = sub * 4 + t;
      v8f c = {0,0,0,0,0,0,0,0};
#pragma unroll
      for (int ks = 0; ks < 8; ++ks)
        c = wmma_bf16(ld_afrag(h2, 264, 0, ks * 32, lane),
                      ld_bfrag(tw2t, 264, nt * 16, ks * 32, lane), c);
      const int col = nt * 16 + (lane & 15);
      const int mb  = (lane < 16) ? 0 : 8;
#pragma unroll
      for (int j = 0; j < 8; ++j) {
        const int gr = rowbase + j + mb;
        g[(size_t)gr * WW + col] = __expf(fast_tanh(c[j] + tb2s[col]));
      }
    }
    // No barrier needed here: next chunk's layer-0 writes only H1, and the
    // barrier before the next layer-1 orders them against partner reads.
  }
}

// ---------------------------------------------------------------------------
// Kernel 3: scattering chain per (b,n,m).
//   res_vs = rw_vs(16x16) @ act(16x128)        -> fp32 WMMA 16x16x4
//   t0     = tanh(res_vs @ sw0^T + sb0)        -> bf16 WMMA 16x16x32
//   scat_v GEMVs in VALU f32; reduce to gdot.
// ---------------------------------------------------------------------------
__global__ __launch_bounds__(256) void k_scat(
    const float* __restrict__ g, const float* __restrict__ scatk,
    const float* __restrict__ selfk, const float* __restrict__ velw,
    const float* __restrict__ sw0, const float* __restrict__ sb0,
    const float* __restrict__ sw1, const float* __restrict__ sb1,
    const float* __restrict__ outw, const __bf16* __restrict__ sw0b,
    float* __restrict__ gdot) {
  __shared__ float act[16 * 128], act2[16 * 128], actv[128];
  __shared__ float rwvs[16 * 16], rwv[16];
  __shared__ float resv0[128], resv1[128], s0s[128], s1s[128], red[128];
  __shared__ __align__(16) __bf16 resvsb[16 * 136];

  const int tid = threadIdx.x;
  const int idx = blockIdx.x;          // == (b*DN+n)*DM + m
  const int n = (idx / DM) % DN;
  const int b = idx / (DM * DN);
  const size_t rbase = (size_t)idx * DK1 * WW;

  for (int i = tid; i < 16 * 128; i += 256) {
    const int k = i >> 7, w = i & 127;
    act[i] = g[rbase + (size_t)(1 + k) * WW + w];
  }
  if (tid < 128) actv[tid] = g[rbase + tid];
  {
    const int i = tid >> 4, j = tid & 15;
    rwvs[tid] = (1.f - selfk[(b * DK + i) * DK + j]) * velw[b * DK + j];
  }
  if (tid < 16)
    rwv[tid] = (1.f - scatk[(b * DN + n) * DK + tid]) * velw[b * DK + tid];
  __syncthreads();

  // scat_v path #1: res_v0 then s0 = tanh(sw0 @ res_v0 + sb0)
  if (tid < 128) {
    float acc = 0.f;
#pragma unroll
    for (int k = 0; k < 16; ++k) acc += rwv[k] * act[k * 128 + tid];
    resv0[tid] = acc;
  }
  __syncthreads();
  if (tid < 128) {
    float acc = sb0[tid];
    const float* rowp = sw0 + tid * 128;
    for (int w = 0; w < 128; ++w) acc += rowp[w] * resv0[w];
    s0s[tid] = fast_tanh(acc);
  }

  const int wv = tid >> 5, lane = tid & 31;
  // GEMM1: res_vs = rw_vs @ act  (fp32 WMMA, 4 K-steps, one N-tile per wave)
  {
    const int mr   = lane & 15;
    const int koff = (lane < 16) ? 0 : 2;
    const int ncol = wv * 16 + (lane & 15);
    v8f c = {0,0,0,0,0,0,0,0};
#pragma unroll
    for (int ks = 0; ks < 4; ++ks) {
      v2f a, bb;
      a[0]  = rwvs[mr * 16 + ks * 4 + koff];
      a[1]  = rwvs[mr * 16 + ks * 4 + koff + 1];
      bb[0] = act[(ks * 4 + koff) * 128 + ncol];
      bb[1] = act[(ks * 4 + koff + 1) * 128 + ncol];
      c = __builtin_amdgcn_wmma_f32_16x16x4_f32(false, a, false, bb,
                                                (short)0, c, false, false);
    }
    const int mb = (lane < 16) ? 0 : 8;
#pragma unroll
    for (int j = 0; j < 8; ++j) resvsb[(j + mb) * 136 + ncol] = (__bf16)c[j];
  }
  __syncthreads();

  // GEMM2: act2 = act + tanh(res_vs @ sw0^T + sb0)  (bf16 WMMA)
  {
    v8f c = {0,0,0,0,0,0,0,0};
#pragma unroll
    for (int ks = 0; ks < 4; ++ks)
      c = wmma_bf16(ld_afrag(resvsb, 136, 0, ks * 32, lane),
                    ld_bfrag(sw0b, 128, wv * 16, ks * 32, lane), c);
    const int col = wv * 16 + (lane & 15);
    const int mb  = (lane < 16) ? 0 : 8;
#pragma unroll
    for (int j = 0; j < 8; ++j) {
      const int r = j + mb;
      act2[r * 128 + col] = act[r * 128 + col] + fast_tanh(c[j] + sb0[col]);
    }
  }
  __syncthreads();

  // scat_v path #2 on updated act_vstar
  if (tid < 128) {
    float acc = 0.f;
#pragma unroll
    for (int k = 0; k < 16; ++k) acc += rwv[k] * act2[k * 128 + tid];
    resv1[tid] = acc;
  }
  __syncthreads();
  if (tid < 128) {
    float acc = sb1[tid];
    const float* rowp = sw1 + tid * 128;
    for (int w = 0; w < 128; ++w) acc += rowp[w] * resv1[w];
    s1s[tid] = fast_tanh(acc);
  }
  __syncthreads();

  // green = act_v + s0 + s1 ;  gdot = green . out_w
  if (tid < 128) red[tid] = (actv[tid] + s0s[tid] + s1s[tid]) * outw[tid];
  __syncthreads();
  for (int s = 64; s > 0; s >>= 1) {
    if (tid < s) red[tid] += red[tid + s];
    __syncthreads();
  }
  if (tid == 0) gdot[idx] = red[0];
}

// ---------------------------------------------------------------------------
// Kernel 4: out[b,n] = sum_m gdot * boundary * boundary_weights
// ---------------------------------------------------------------------------
__global__ __launch_bounds__(128) void k_out(
    const float* __restrict__ gdot, const float* __restrict__ boundary,
    const float* __restrict__ bw, float* __restrict__ out) {
  const int t = threadIdx.x;  // t = b*DN + n
  const int b = t >> 5;
  float acc = 0.f;
  for (int m = 0; m < DM; ++m)
    acc += gdot[t * DM + m] * boundary[b * DM + m] * bw[b * DM + m];
  out[t] = acc;
}

// ---------------------------------------------------------------------------
extern "C" void kernel_launch(void* const* d_in, const int* in_sizes, int n_in,
                              void* d_out, int out_size, void* d_ws,
                              size_t ws_size, hipStream_t stream) {
  const float* phase    = (const float*)d_in[0];
  const float* bco      = (const float*)d_in[1];
  const float* boundary = (const float*)d_in[2];
  const float* bwt      = (const float*)d_in[3];
  const float* posc     = (const float*)d_in[4];
  const float* sigma    = (const float*)d_in[5];
  const float* velc     = (const float*)d_in[6];
  const float* velw     = (const float*)d_in[7];
  const float* scatk    = (const float*)d_in[8];
  const float* selfk    = (const float*)d_in[9];
  const float* aw0 = (const float*)d_in[10];
  const float* ab0 = (const float*)d_in[11];
  const float* aw1 = (const float*)d_in[12];
  const float* ab1 = (const float*)d_in[13];
  const float* aw2 = (const float*)d_in[14];
  const float* ab2 = (const float*)d_in[15];
  const float* tw0 = (const float*)d_in[16];
  const float* tb0 = (const float*)d_in[17];
  const float* tw1 = (const float*)d_in[18];
  const float* tb1 = (const float*)d_in[19];
  const float* tw2 = (const float*)d_in[20];
  const float* tb2 = (const float*)d_in[21];
  const float* sw0 = (const float*)d_in[22];
  const float* sb0 = (const float*)d_in[23];
  const float* sw1 = (const float*)d_in[24];
  const float* sb1 = (const float*)d_in[25];
  const float* outw = (const float*)d_in[26];

  char* ws = (char*)d_ws;
  float*  coeff = (float*)(ws + 0);            // 4352 f32
  float*  gdot  = (float*)(ws + 32768);        // 8192 f32
  __bf16* sw0b  = (__bf16*)(ws + 65536);       // 16384 bf16
  float*  g     = (float*)(ws + 131072);       // 139264*128 f32 (~71 MB)

  k_prep<<<64, 256, 0, stream>>>(sw0, sw0b);
  k_attn<<<DB * DN * DK1, 256, 0, stream>>>(phase, posc, sigma, velc, aw0, ab0,
                                            aw1, ab1, aw2, ab2, coeff);
  k_tmlp<<<512, 256, 0, stream>>>(phase, bco, velc, coeff, tw0, tb0, tw1, tb1,
                                  tw2, tb2, g);
  k_scat<<<DB * DN * DM, 256, 0, stream>>>(g, scatk, selfk, velw, sw0, sb0,
                                           sw1, sb1, outw, sw0b, gdot);
  k_out<<<1, 128, 0, stream>>>(gdot, boundary, bwt, (float*)d_out);
}
